// FasterRCNN_4492535792539
// MI455X (gfx1250) — matrix operand order repaired and checked by reading, not compiled, not run on hardware
//
#include <hip/hip_runtime.h>
#include <hip/hip_bf16.h>

// ---------------- problem constants (from reference) ----------------
#define NROI   128
#define NCH    2048
#define HF     16
#define WF     16
#define POOL   7
#define KTOT   (NCH * POOL * POOL)   // 100352
#define NHID   1024
#define NLOC   324
#define NCLS   81
#define SPLITK 8
#define KPER   (KTOT / SPLITK)       // 12544 = 98 * 128
#define KSTEP  128
#define NEGV   (-1e30f)

typedef __attribute__((ext_vector_type(16))) __bf16 v16bf;
typedef __attribute__((ext_vector_type(8)))  __bf16 v8bf;
typedef __attribute__((ext_vector_type(4)))  __bf16 v4bf;
typedef __attribute__((ext_vector_type(8)))  float  v8f;
typedef __attribute__((ext_vector_type(4)))  float  v4f;

static __device__ __forceinline__ __bf16 f2bf(float f) {
    union { float f; unsigned u; } a; a.f = f;
    unsigned r = (a.u + 0x7FFFu + ((a.u >> 16) & 1u)) >> 16;   // RNE
    union { unsigned short s; __bf16 b; } o; o.s = (unsigned short)r;
    return o.b;
}

static __device__ __forceinline__ v4bf cvt4(v4f f) {
    v4bf b;
    b[0] = f2bf(f[0]); b[1] = f2bf(f[1]); b[2] = f2bf(f[2]); b[3] = f2bf(f[3]);
    return b;
}

// ---------------- kernel 1: greedy NMS (torchvision semantics) ----------------
__global__ void nms_kernel(const float* __restrict__ rois,
                           const float* __restrict__ rpn,
                           const int*   __restrict__ lbl,
                           int* __restrict__ keepmask,
                           float* __restrict__ outKeep) {
    const int t = threadIdx.x;                 // 0..127
    __shared__ float  ss[NROI];
    __shared__ float4 sb[NROI];
    __shared__ int    sorig[NROI];
    __shared__ int    svalid[NROI];
    __shared__ int    skeep[NROI];

    float4 box = make_float4(rois[t*4+0], rois[t*4+1], rois[t*4+2], rois[t*4+3]);
    int valid = (lbl[t] != -1) ? 1 : 0;
    float sc = valid ? rpn[t*2+1] : -INFINITY;
    ss[t] = sc; svalid[t] = valid; skeep[t] = 1;
    __syncthreads();

    // stable descending rank (ties broken by original index)
    int rank = 0;
    for (int j = 0; j < NROI; ++j) {
        float sj = ss[j];
        if (sj > sc || (sj == sc && j < t)) ++rank;
    }
    sb[rank] = box; sorig[rank] = t;
    __syncthreads();

    float4 mb = sb[t];                          // box at sorted position t
    float myArea = (mb.z - mb.x) * (mb.w - mb.y);
    for (int i = 0; i < NROI - 1; ++i) {
        __syncthreads();
        if (t > i && skeep[t] && skeep[i]) {
            float4 bi = sb[i];
            float ltx = fmaxf(bi.x, mb.x), lty = fmaxf(bi.y, mb.y);
            float rbx = fminf(bi.z, mb.z), rby = fminf(bi.w, mb.w);
            float iw = fmaxf(rbx - ltx, 0.f), ih = fmaxf(rby - lty, 0.f);
            float inter = iw * ih;
            float ai = (bi.z - bi.x) * (bi.w - bi.y);
            float iou = inter / (ai + myArea - inter);
            if (iou > 0.6f) skeep[t] = 0;
        }
    }
    __syncthreads();
    int orig = sorig[t];
    int kv = skeep[t] && svalid[orig];
    keepmask[orig] = kv;
    outKeep[orig] = kv ? 1.0f : 0.0f;
}

// ---------------- kernel 2: ROI adaptive max-pool -> bf16 activations ----------------
__global__ void roipool_kernel(const float* __restrict__ feat,
                               const float* __restrict__ rois,
                               __bf16* __restrict__ pooled) {
    int gid = blockIdx.x * blockDim.x + threadIdx.x;
    if (gid >= NROI * NCH * POOL * POOL) return;
    int n    = gid / (NCH * POOL * POOL);
    int rem  = gid % (NCH * POOL * POOL);
    int c    = rem / (POOL * POOL);
    int cell = rem % (POOL * POOL);
    int i = cell / POOL, j = cell % POOL;

    float cx = floorf(rois[n*4+0] + 0.5f);
    float cy = floorf(rois[n*4+1] + 0.5f);
    float w  = floorf(rois[n*4+2] + 0.5f);
    float h  = floorf(rois[n*4+3] + 0.5f);
    int x1 = (int)floorf(fminf(fmaxf((cx - w*0.5f) * (1.f/32.f), 0.f), (float)(WF-1)));
    int y1 = (int)floorf(fminf(fmaxf((cy - h*0.5f) * (1.f/32.f), 0.f), (float)(HF-1)));
    int x2 = (int)floorf(fminf(fmaxf((cx + w*0.5f) * (1.f/32.f), 1.f), (float)WF));
    int y2 = (int)floorf(fminf(fmaxf((cy + h*0.5f) * (1.f/32.f), 1.f), (float)HF));
    x1 = min(x1, x2 - 1); y1 = min(y1, y2 - 1);
    int H = y2 - y1, W = x2 - x1;
    int ys = y1 + (i * H) / POOL,  ye = y1 + ((i + 1) * H + POOL - 1) / POOL;
    int xs = x1 + (j * W) / POOL,  xe = x1 + ((j + 1) * W + POOL - 1) / POOL;

    const float* fch = feat + (size_t)c * (HF * WF);
    float m = NEGV;
    #pragma unroll
    for (int a = 0; a < 4; ++a) {
        int ry = ys + a;
        if (ry < ye) {
            #pragma unroll
            for (int b = 0; b < 4; ++b) {
                int rx = xs + b;
                if (rx < xe) m = fmaxf(m, fch[ry * WF + rx]);
            }
        }
    }
    pooled[(size_t)n * KTOT + c * (POOL*POOL) + cell] = f2bf(m);
}

// ---------------- kernel 3: FC GEMM via v_wmma_f32_16x16x32_bf16 ----------------
// grid = (64 column-blocks, SPLITK=8); block = 256 threads = 8 waves.
// Wave w computes the 16x16 tile  rows[16w..16w+15] x cols[colbase..colbase+15].
// fc_w streamed exactly once (NT loads, fp32 -> bf16), double-buffered in LDS:
// one workgroup barrier per 128-k step, 4 WMMAs per barrier.
__global__ __launch_bounds__(256)
void fc_wmma_kernel(const __bf16* __restrict__ pooled,
                    const float*  __restrict__ fcw,
                    float* __restrict__ partial) {
    __shared__ __align__(16) __bf16 ldsB[2][16 * KSTEP];   // 2 x 4 KB double buffer

    const int colbase = blockIdx.x * 16;
    const int split   = blockIdx.y;
    const int t    = threadIdx.x;
    const int wave = t >> 5;
    const int lane = t & 31;
    const int half = (lane >> 4) & 1;       // lane group 0/1
    const int nm   = lane & 15;             // m for A-frag, n for B-frag

    // staging: thread t loads 2x4 floats of fc_w row (t>>4), chunk (t&15)
    const int ldRow = t >> 4, ldChunk = t & 15;
    const float*  gB = fcw + (size_t)(colbase + ldRow) * KTOT
                           + (size_t)split * KPER + ldChunk * 4;
    // A-frag base: row = wave*16 + nm; K-contiguous layout uses half*16
    const __bf16* gA = pooled + (size_t)(wave * 16 + nm) * KTOT
                              + (size_t)split * KPER + half * 16;
    __bf16*       ldsW = &ldsB[0][ldRow * KSTEP + ldChunk * 4];
    const __bf16* ldsR = &ldsB[0][nm * KSTEP + half * 16];

    // prologue: stage tile 0 into buffer 0
    {
        v4bf b0 = cvt4(__builtin_nontemporal_load((const v4f*)(gB)));
        v4bf b1 = cvt4(__builtin_nontemporal_load((const v4f*)(gB + 64)));
        *(v4bf*)(ldsW)      = b0;
        *(v4bf*)(ldsW + 64) = b1;
    }
    __syncthreads();

    v8f acc = {};
    int cur = 0;
    for (int k0 = 0; k0 < KPER; k0 += KSTEP) {
        const bool more = (k0 + KSTEP) < KPER;
        // issue next fc_w tile early (non-temporal stream), hide HBM latency under WMMA
        v4f f0, f1;
        if (more) {
            f0 = __builtin_nontemporal_load((const v4f*)(gB + k0 + KSTEP));
            f1 = __builtin_nontemporal_load((const v4f*)(gB + k0 + KSTEP + 64));
            __builtin_prefetch(gB + k0 + 2 * KSTEP, 0, 1);   // global_prefetch_b8
        }

        const __bf16* lb = ldsR + cur * (16 * KSTEP);
        #pragma unroll
        for (int sub = 0; sub < 4; ++sub) {
            // A 16x32 bf16 layout: lane(half,m) holds K = half*16 + e (K-contiguous)
            const __bf16* ga = gA + k0 + sub * 32;
            v8bf alo = *(const v8bf*)(ga);
            v8bf ahi = *(const v8bf*)(ga + 8);
            v16bf a = __builtin_shufflevector(alo, ahi,
                        0,1,2,3,4,5,6,7,8,9,10,11,12,13,14,15);
            // B 32x16 bf16 layout: lane(half,n) holds K = half*16 + e (K-contiguous)
            v16bf b = *(const v16bf*)(lb + sub * 32);
            acc = __builtin_amdgcn_wmma_f32_16x16x32_bf16(false, a, false, b,
                                                          (short)0, acc, false, false);
        }

        if (more) {
            __bf16* lw = ldsW + (cur ^ 1) * (16 * KSTEP);
            *(v4bf*)(lw)      = cvt4(f0);
            *(v4bf*)(lw + 64) = cvt4(f1);
        }
        __syncthreads();        // single barrier per 128-k step (double buffered)
        cur ^= 1;
    }

    // D layout: VGPR r -> M = r + 8*half ; N = nm
    float* out = partial + (size_t)split * NROI * NHID
               + (size_t)(wave * 16 + half * 8) * NHID + colbase + nm;
    #pragma unroll
    for (int r = 0; r < 8; ++r) out[r * NHID] = acc[r];
}

// ---------------- kernel 4: deterministic split-K reduce + bias ----------------
__global__ void reduce_bias_kernel(const float* __restrict__ partial,
                                   const float* __restrict__ fcb,
                                   float* __restrict__ hid) {
    int t = blockIdx.x * blockDim.x + threadIdx.x;
    if (t >= NROI * NHID) return;
    int c = t % NHID;
    float s = fcb[c];
    #pragma unroll
    for (int sp = 0; sp < SPLITK; ++sp) s += partial[(size_t)sp * NROI * NHID + t];
    hid[t] = s;
}

// ---------------- kernel 5: cls/loc heads + keep masking ----------------
__global__ void heads_kernel(const float* __restrict__ hid,
                             const float* __restrict__ clsw, const float* __restrict__ clsb,
                             const float* __restrict__ locw, const float* __restrict__ locb,
                             const int* __restrict__ keepmask,
                             float* __restrict__ out) {
    int t = blockIdx.x * blockDim.x + threadIdx.x;
    if (t >= NROI * (NLOC + NCLS)) return;
    int n = t / (NLOC + NCLS), o = t % (NLOC + NCLS);
    const float* hrow = hid + (size_t)n * NHID;
    const float* wrow; float bias; float* dst;
    if (o < NLOC) { wrow = locw + (size_t)o * NHID;          bias = locb[o];
                    dst = out + (size_t)n * NLOC + o; }
    else          { int q = o - NLOC;
                    wrow = clsw + (size_t)q * NHID;          bias = clsb[q];
                    dst = out + (size_t)NROI * NLOC + (size_t)n * NCLS + q; }
    float s = bias;
    for (int k = 0; k < NHID; k += 4) {
        v4f hv = *(const v4f*)(hrow + k);
        v4f wv = *(const v4f*)(wrow + k);
        s += hv[0]*wv[0] + hv[1]*wv[1] + hv[2]*wv[2] + hv[3]*wv[3];
    }
    *dst = keepmask[n] ? s : 0.0f;
}

// ---------------- launch ----------------
extern "C" void kernel_launch(void* const* d_in, const int* in_sizes, int n_in,
                              void* d_out, int out_size, void* d_ws, size_t ws_size,
                              hipStream_t stream) {
    const float* feat = (const float*)d_in[0];   // [1,2048,16,16]
    const float* rois = (const float*)d_in[1];   // [128,4]
    const float* rpn  = (const float*)d_in[2];   // [128,2]
    const int*   lbl  = (const int*)  d_in[3];   // [128]
    const float* fcw  = (const float*)d_in[4];   // [1024,100352]
    const float* fcb  = (const float*)d_in[5];   // [1024]
    const float* clsw = (const float*)d_in[6];   // [81,1024]
    const float* clsb = (const float*)d_in[7];   // [81]
    const float* locw = (const float*)d_in[8];   // [324,1024]
    const float* locb = (const float*)d_in[9];   // [324]
    float* out = (float*)d_out;                  // [128*324 | 128*81 | 128]

    // workspace layout (all 16B aligned)
    char* ws = (char*)d_ws;
    const size_t POOLED_B  = (size_t)NROI * KTOT * 2;            // 25,690,112
    const size_t PARTIAL_B = (size_t)SPLITK * NROI * NHID * 4;   //  4,194,304
    const size_t HID_B     = (size_t)NROI * NHID * 4;            //    524,288
    __bf16* pooled   = (__bf16*)ws;
    float*  partial  = (float*)(ws + POOLED_B);
    float*  hid      = (float*)(ws + POOLED_B + PARTIAL_B);
    int*    keepmask = (int*)  (ws + POOLED_B + PARTIAL_B + HID_B);

    nms_kernel<<<1, NROI, 0, stream>>>(rois, rpn, lbl, keepmask,
                                       out + (size_t)NROI * (NLOC + NCLS));

    {
        int total = NROI * NCH * POOL * POOL;
        roipool_kernel<<<(total + 255) / 256, 256, 0, stream>>>(feat, rois, pooled);
    }

    fc_wmma_kernel<<<dim3(NHID / 16, SPLITK), 256, 0, stream>>>(pooled, fcw, partial);

    {
        int total = NROI * NHID;
        reduce_bias_kernel<<<(total + 255) / 256, 256, 0, stream>>>(partial, fcb, hid);
    }
    {
        int total = NROI * (NLOC + NCLS);
        heads_kernel<<<(total + 255) / 256, 256, 0, stream>>>(hid, clsw, clsb,
                                                              locw, locb, keepmask, out);
    }
}
